// SymbolsDecoder_21208548508123
// MI455X (gfx1250) — compile-verified, matrix operand order checked by reading, not run on hardware
//
#include <hip/hip_runtime.h>

#define BDIM 1024
#define LDIM 80
#define EDIM 256
#define DDIM 256
#define TSTEPS 20
#define NSPEC 4
#define NSYMB 100

typedef __attribute__((ext_vector_type(16))) __bf16 v16bf;
typedef __attribute__((ext_vector_type(8)))  __bf16 v8bf;
typedef __attribute__((ext_vector_type(8)))  float  v8f;

__device__ __forceinline__ float sigmoidf_(float x) { return 1.0f / (1.0f + __expf(-x)); }

// Load a 16x32 bf16 A/B fragment for v_wmma_f32_16x16x32_bf16 from a row-major
// matrix (rows = M for A, rows = N for B since we compute X @ W^T with W[N,K]).
// ISA layout: lanes 0-15 hold row (lane), K = kb+{0..7} and kb+{16..23};
// lanes 16-31 hold the same rows with K = kb+{8..15} and kb+{24..31}.
__device__ __forceinline__ v16bf frag_ld(const __bf16* __restrict__ base, int row0, int kb, int ld) {
    const int lane = threadIdx.x & 31;
    const int r = lane & 15;
    const int k0 = kb + ((lane & 16) ? 8 : 0);
    const __bf16* p = base + (size_t)(row0 + r) * ld + k0;
    union { v16bf v; v8bf h[2]; } u;
    u.h[0] = *(const v8bf*)(p);
    u.h[1] = *(const v8bf*)(p + 16);
    return u.v;
}

__device__ __forceinline__ v8f wmma_bf16(v16bf a, v16bf b, v8f c) {
    return __builtin_amdgcn_wmma_f32_16x16x32_bf16(false, a, false, b, (short)0, c, false, false);
}

#define V8F_ZERO {0.f,0.f,0.f,0.f,0.f,0.f,0.f,0.f}

// ---------------------------------------------------------------- setup kernels

__global__ void k_cvt_bf16(const float* __restrict__ src, __bf16* __restrict__ dst, int n) {
    int i = blockIdx.x * blockDim.x + threadIdx.x;
    if (i < n) dst[i] = (__bf16)src[i];
}

// emb[T,B,D] gather: idx<NS -> common table, else per-example symbol encodings.
__global__ void k_embed(const float* __restrict__ common, const float* __restrict__ symenc,
                        const int* __restrict__ idx, __bf16* __restrict__ embT) {
    int gid = blockIdx.x * 256 + threadIdx.x;     // (t*B + b)*D + e
    int e  = gid & (DDIM - 1);
    int bt = gid >> 8;
    int b  = bt % BDIM;
    int t  = bt / BDIM;
    int ix = idx[b * TSTEPS + t];
    float v;
    if (ix < NSPEC) {
        int c = ix < 0 ? 0 : ix;
        v = common[c * DDIM + e];
    } else {
        int s = ix - NSPEC;
        if (s > NSYMB - 1) s = NSYMB - 1;
        v = symenc[((size_t)b * NSYMB + s) * DDIM + e];
    }
    embT[gid] = (__bf16)v;
}

// ---------------------------------------------------------------- GEMM kernels
// One wave per C macro-tile; 2x M register blocking amortizes B(weight) loads.

// aw[B,80] = [emb | h1] @ W_attn^T + b_attn   (K=512 over 2 A srcs; 32xM per wave)
__global__ void k_attn_gemm(const __bf16* __restrict__ Aemb, const __bf16* __restrict__ Ah1,
                            const __bf16* __restrict__ Wa, const float* __restrict__ bias,
                            float* __restrict__ aw) {
    const int m0 = blockIdx.x * 32, n0 = blockIdx.y * 16;
    v8f acc[2] = {V8F_ZERO, V8F_ZERO};
    for (int kb = 0; kb < DDIM; kb += 32) {
        v16bf b = frag_ld(Wa, n0, kb, 2 * DDIM);
        acc[0] = wmma_bf16(frag_ld(Aemb, m0,      kb, DDIM), b, acc[0]);
        acc[1] = wmma_bf16(frag_ld(Aemb, m0 + 16, kb, DDIM), b, acc[1]);
    }
    for (int kb = 0; kb < DDIM; kb += 32) {
        v16bf b = frag_ld(Wa, n0, DDIM + kb, 2 * DDIM);
        acc[0] = wmma_bf16(frag_ld(Ah1, m0,      kb, DDIM), b, acc[0]);
        acc[1] = wmma_bf16(frag_ld(Ah1, m0 + 16, kb, DDIM), b, acc[1]);
    }
    const int lane = threadIdx.x & 31;
    const int n = n0 + (lane & 15);
    const int mh = (lane & 16) ? 8 : 0;
    const float bn = bias[n];
#pragma unroll
    for (int mt = 0; mt < 2; ++mt)
#pragma unroll
        for (int i = 0; i < 8; ++i)
            aw[(size_t)(m0 + mt * 16 + mh + i) * LDIM + n] = acc[mt][i] + bn;
}

// per-row softmax over L=80 then context: applied[b,e] = sum_l p[l]*enc[b,l,e]
__global__ void k_softmax_ctx(const float* __restrict__ aw, const float* __restrict__ enc,
                              __bf16* __restrict__ applied) {
    __shared__ float sm[256];
    __shared__ float p[LDIM];
    const int b = blockIdx.x, tid = threadIdx.x;
    float v = (tid < LDIM) ? aw[(size_t)b * LDIM + tid] : -3.4e38f;
    sm[tid] = v; __syncthreads();
    for (int s = 128; s > 0; s >>= 1) { if (tid < s) sm[tid] = fmaxf(sm[tid], sm[tid + s]); __syncthreads(); }
    float mx = sm[0]; __syncthreads();
    float ev = (tid < LDIM) ? __expf(v - mx) : 0.0f;
    sm[tid] = ev; __syncthreads();
    for (int s = 128; s > 0; s >>= 1) { if (tid < s) sm[tid] += sm[tid + s]; __syncthreads(); }
    float inv = 1.0f / sm[0];
    if (tid < LDIM) p[tid] = ev * inv;
    __syncthreads();
    const float* eb = enc + (size_t)b * LDIM * EDIM + tid;   // coalesced over e = tid
    float s = 0.0f;
#pragma unroll 8
    for (int l = 0; l < LDIM; ++l) s += p[l] * eb[(size_t)l * EDIM];
    applied[(size_t)b * EDIM + tid] = (__bf16)s;
}

// comb = relu([emb | applied] @ W_comb^T + b_comb), bf16 out; 2x2 register tile.
__global__ void k_comb_gemm(const __bf16* __restrict__ Aemb, const __bf16* __restrict__ Aapp,
                            const __bf16* __restrict__ Wc, const float* __restrict__ bias,
                            __bf16* __restrict__ comb) {
    const int m0 = blockIdx.x * 32, n0 = blockIdx.y * 32;
    v8f acc[2][2] = {{V8F_ZERO, V8F_ZERO}, {V8F_ZERO, V8F_ZERO}};
    for (int kb = 0; kb < DDIM; kb += 32) {
        v16bf a0 = frag_ld(Aemb, m0,      kb, DDIM);
        v16bf a1 = frag_ld(Aemb, m0 + 16, kb, DDIM);
        v16bf b0 = frag_ld(Wc, n0,      kb, 2 * DDIM);
        v16bf b1 = frag_ld(Wc, n0 + 16, kb, 2 * DDIM);
        acc[0][0] = wmma_bf16(a0, b0, acc[0][0]);
        acc[0][1] = wmma_bf16(a0, b1, acc[0][1]);
        acc[1][0] = wmma_bf16(a1, b0, acc[1][0]);
        acc[1][1] = wmma_bf16(a1, b1, acc[1][1]);
    }
    for (int kb = 0; kb < DDIM; kb += 32) {
        v16bf a0 = frag_ld(Aapp, m0,      kb, DDIM);
        v16bf a1 = frag_ld(Aapp, m0 + 16, kb, DDIM);
        v16bf b0 = frag_ld(Wc, n0,      DDIM + kb, 2 * DDIM);
        v16bf b1 = frag_ld(Wc, n0 + 16, DDIM + kb, 2 * DDIM);
        acc[0][0] = wmma_bf16(a0, b0, acc[0][0]);
        acc[0][1] = wmma_bf16(a0, b1, acc[0][1]);
        acc[1][0] = wmma_bf16(a1, b0, acc[1][0]);
        acc[1][1] = wmma_bf16(a1, b1, acc[1][1]);
    }
    const int lane = threadIdx.x & 31;
    const int nl = lane & 15;
    const int mh = (lane & 16) ? 8 : 0;
#pragma unroll
    for (int mt = 0; mt < 2; ++mt)
#pragma unroll
        for (int nt = 0; nt < 2; ++nt) {
            const int n = n0 + nt * 16 + nl;
            const float bn = bias[n];
#pragma unroll
            for (int i = 0; i < 8; ++i) {
                float v = acc[mt][nt][i] + bn;
                v = v > 0.f ? v : 0.f;
                comb[(size_t)(m0 + mt * 16 + mh + i) * DDIM + n] = (__bf16)v;
            }
        }
}

// Fused LSTM cell: wave owns 32 batch rows x 16 hidden dims for all 4 gates
// (columns d0 + g*256 of the [B,1024] gate matrix); cell update in epilogue,
// c in place, h written as bf16 into the ping-pong buffer.
__global__ void k_lstm(const __bf16* __restrict__ X, const __bf16* __restrict__ Hprev,
                       const __bf16* __restrict__ Wih, const __bf16* __restrict__ Whh,
                       const float* __restrict__ bih, const float* __restrict__ bhh,
                       float* __restrict__ c, __bf16* __restrict__ Hout) {
    const int m0 = blockIdx.x * 32, d0 = blockIdx.y * 16;
    v8f acc[2][4] = {{V8F_ZERO, V8F_ZERO, V8F_ZERO, V8F_ZERO},
                     {V8F_ZERO, V8F_ZERO, V8F_ZERO, V8F_ZERO}};
    for (int kb = 0; kb < DDIM; kb += 32) {
        v16bf a0 = frag_ld(X, m0,      kb, DDIM);
        v16bf a1 = frag_ld(X, m0 + 16, kb, DDIM);
#pragma unroll
        for (int g = 0; g < 4; ++g) {
            v16bf w = frag_ld(Wih, g * DDIM + d0, kb, DDIM);
            acc[0][g] = wmma_bf16(a0, w, acc[0][g]);
            acc[1][g] = wmma_bf16(a1, w, acc[1][g]);
        }
    }
    for (int kb = 0; kb < DDIM; kb += 32) {
        v16bf a0 = frag_ld(Hprev, m0,      kb, DDIM);
        v16bf a1 = frag_ld(Hprev, m0 + 16, kb, DDIM);
#pragma unroll
        for (int g = 0; g < 4; ++g) {
            v16bf w = frag_ld(Whh, g * DDIM + d0, kb, DDIM);
            acc[0][g] = wmma_bf16(a0, w, acc[0][g]);
            acc[1][g] = wmma_bf16(a1, w, acc[1][g]);
        }
    }
    const int lane = threadIdx.x & 31;
    const int n = d0 + (lane & 15);
    const int mh = (lane & 16) ? 8 : 0;
    const float b_i = bih[n]            + bhh[n];
    const float b_f = bih[DDIM + n]     + bhh[DDIM + n];
    const float b_g = bih[2 * DDIM + n] + bhh[2 * DDIM + n];
    const float b_o = bih[3 * DDIM + n] + bhh[3 * DDIM + n];
#pragma unroll
    for (int mt = 0; mt < 2; ++mt) {
#pragma unroll
        for (int i = 0; i < 8; ++i) {
            size_t off = (size_t)(m0 + mt * 16 + mh + i) * DDIM + n;
            float gi = acc[mt][0][i] + b_i;
            float gf = acc[mt][1][i] + b_f;
            float gg = acc[mt][2][i] + b_g;
            float go = acc[mt][3][i] + b_o;
            float cn = sigmoidf_(gf) * c[off] + sigmoidf_(gi) * tanhf(gg);
            float h  = sigmoidf_(go) * tanhf(cn);
            c[off] = cn;
            Hout[off] = (__bf16)h;
        }
    }
}

// Fused output projection + log_softmax: 128-thread block (4 waves) computes a
// full 16x256 logits tile (wave w -> N tiles w*64..w*64+63) into LDS, then does
// the row-wise log-softmax in-block and writes out[t] directly.
__global__ void __launch_bounds__(128) k_out_lsm(const __bf16* __restrict__ H1,
                                                 const __bf16* __restrict__ Wo,
                                                 const float* __restrict__ bias,
                                                 float* __restrict__ outT) {
    __shared__ float sm[16][DDIM + 4];   // +4 pad: kills bank conflicts on row scans
    __shared__ float pm[16][8], ps[16][8];
    const int tid = threadIdx.x;
    const int lane = tid & 31, wave = tid >> 5;
    const int m0 = blockIdx.x * 16;
    const int nw = wave * 64;
    v8f acc[4] = {V8F_ZERO, V8F_ZERO, V8F_ZERO, V8F_ZERO};
    for (int kb = 0; kb < DDIM; kb += 32) {
        v16bf a = frag_ld(H1, m0, kb, DDIM);
#pragma unroll
        for (int j = 0; j < 4; ++j)
            acc[j] = wmma_bf16(a, frag_ld(Wo, nw + j * 16, kb, DDIM), acc[j]);
    }
    const int nl = lane & 15;
    const int mh = (lane & 16) ? 8 : 0;
#pragma unroll
    for (int j = 0; j < 4; ++j) {
        const int n = nw + j * 16 + nl;
        const float bn = bias[n];
#pragma unroll
        for (int i = 0; i < 8; ++i) sm[mh + i][n] = acc[j][i] + bn;
    }
    __syncthreads();
    // 8 threads per row: partial max / sum over 32-col strips, combine, emit.
    const int row = tid & 15, sub = tid >> 4;            // sub in 0..7
    float mx = -3.4e38f;
    for (int cc = sub * 32; cc < sub * 32 + 32; ++cc) mx = fmaxf(mx, sm[row][cc]);
    pm[row][sub] = mx; __syncthreads();
    if (sub == 0) {
        float m2 = pm[row][0];
        for (int k = 1; k < 8; ++k) m2 = fmaxf(m2, pm[row][k]);
        pm[row][0] = m2;
    }
    __syncthreads();
    mx = pm[row][0];
    float s = 0.f;
    for (int cc = sub * 32; cc < sub * 32 + 32; ++cc) s += __expf(sm[row][cc] - mx);
    ps[row][sub] = s; __syncthreads();
    if (sub == 0) {
        float s2 = 0.f;
        for (int k = 0; k < 8; ++k) s2 += ps[row][k];
        ps[row][0] = s2;
    }
    __syncthreads();
    const float lse = logf(ps[row][0]) + mx;
    float* orow = outT + (size_t)(m0 + row) * DDIM;
    for (int cc = sub * 32; cc < sub * 32 + 32; ++cc) orow[cc] = sm[row][cc] - lse;
}

// ---------------------------------------------------------------- launcher

extern "C" void kernel_launch(void* const* d_in, const int* in_sizes, int n_in,
                              void* d_out, int out_size, void* d_ws, size_t ws_size,
                              hipStream_t stream) {
    (void)in_sizes; (void)n_in; (void)out_size; (void)ws_size;
    const float* enc    = (const float*)d_in[0];   // [B,L,E]
    const float* symenc = (const float*)d_in[1];   // [B,NSYM,D]
    const int*   idx    = (const int*)  d_in[2];   // [B,T]
    const float* common = (const float*)d_in[3];   // [NS,D]
    const float* W_attn = (const float*)d_in[4];   // [L,2D]
    const float* b_attn = (const float*)d_in[5];
    const float* W_comb = (const float*)d_in[6];   // [D,2D]
    const float* b_comb = (const float*)d_in[7];
    const float* Wih0   = (const float*)d_in[8];   // [4D,D]
    const float* Whh0   = (const float*)d_in[9];
    const float* bih0   = (const float*)d_in[10];
    const float* bhh0   = (const float*)d_in[11];
    const float* Wih1   = (const float*)d_in[12];
    const float* Whh1   = (const float*)d_in[13];
    const float* bih1   = (const float*)d_in[14];
    const float* bhh1   = (const float*)d_in[15];
    const float* W_out  = (const float*)d_in[16];  // [D,E]
    const float* b_out  = (const float*)d_in[17];
    float* out = (float*)d_out;                    // [T,B,D]

    char* wp = (char*)d_ws;
    auto carve = [&](size_t bytes) -> char* {
        char* r = wp; wp += (bytes + 255) & ~(size_t)255; return r;
    };
    __bf16* embT  = (__bf16*)carve((size_t)TSTEPS * BDIM * DDIM * 2);
    __bf16* WaB   = (__bf16*)carve((size_t)LDIM * 2 * DDIM * 2);
    __bf16* WcB   = (__bf16*)carve((size_t)DDIM * 2 * DDIM * 2);
    __bf16* Wih0B = (__bf16*)carve((size_t)4 * DDIM * DDIM * 2);
    __bf16* Whh0B = (__bf16*)carve((size_t)4 * DDIM * DDIM * 2);
    __bf16* Wih1B = (__bf16*)carve((size_t)4 * DDIM * DDIM * 2);
    __bf16* Whh1B = (__bf16*)carve((size_t)4 * DDIM * DDIM * 2);
    __bf16* WoB   = (__bf16*)carve((size_t)DDIM * DDIM * 2);
    __bf16* h0b[2] = { (__bf16*)carve((size_t)BDIM * DDIM * 2),
                       (__bf16*)carve((size_t)BDIM * DDIM * 2) };
    __bf16* h1b[2] = { (__bf16*)carve((size_t)BDIM * DDIM * 2),
                       (__bf16*)carve((size_t)BDIM * DDIM * 2) };
    __bf16* combB = (__bf16*)carve((size_t)BDIM * DDIM * 2);
    __bf16* appB  = (__bf16*)carve((size_t)BDIM * DDIM * 2);
    float* c0     = (float*)carve((size_t)BDIM * DDIM * 4);
    float* c1     = (float*)carve((size_t)BDIM * DDIM * 4);
    float* aw     = (float*)carve((size_t)BDIM * LDIM * 4);

    const int cb = 256;
    auto cvt = [&](const float* s, __bf16* d, int n) {
        k_cvt_bf16<<<(n + cb - 1) / cb, cb, 0, stream>>>(s, d, n);
    };
    cvt(W_attn, WaB,   LDIM * 2 * DDIM);
    cvt(W_comb, WcB,   DDIM * 2 * DDIM);
    cvt(Wih0,   Wih0B, 4 * DDIM * DDIM);
    cvt(Whh0,   Whh0B, 4 * DDIM * DDIM);
    cvt(Wih1,   Wih1B, 4 * DDIM * DDIM);
    cvt(Whh1,   Whh1B, 4 * DDIM * DDIM);
    cvt(W_out,  WoB,   DDIM * DDIM);
    k_embed<<<(TSTEPS * BDIM * DDIM) / 256, 256, 0, stream>>>(common, symenc, idx, embT);
    hipMemsetAsync(c0, 0, (size_t)BDIM * DDIM * 4, stream);
    hipMemsetAsync(c1, 0, (size_t)BDIM * DDIM * 4, stream);
    hipMemsetAsync(h0b[0], 0, (size_t)BDIM * DDIM * 2, stream);
    hipMemsetAsync(h1b[0], 0, (size_t)BDIM * DDIM * 2, stream);

    const dim3 wblk(32);
    for (int t = 0; t < TSTEPS; ++t) {
        const __bf16* embt = embT + (size_t)t * BDIM * DDIM;
        const int p = t & 1, q = p ^ 1;   // read prev from [p], write cur to [q]
        k_attn_gemm<<<dim3(BDIM / 32, LDIM / 16), wblk, 0, stream>>>(embt, h1b[p], WaB, b_attn, aw);
        k_softmax_ctx<<<BDIM, 256, 0, stream>>>(aw, enc, appB);
        k_comb_gemm<<<dim3(BDIM / 32, DDIM / 32), wblk, 0, stream>>>(embt, appB, WcB, b_comb, combB);
        k_lstm<<<dim3(BDIM / 32, DDIM / 16), wblk, 0, stream>>>(combB, h0b[p], Wih0B, Whh0B,
                                                                bih0, bhh0, c0, h0b[q]);
        k_lstm<<<dim3(BDIM / 32, DDIM / 16), wblk, 0, stream>>>(h0b[q], h1b[p], Wih1B, Whh1B,
                                                                bih1, bhh1, c1, h1b[q]);
        k_out_lsm<<<dim3(BDIM / 16), dim3(128), 0, stream>>>(h1b[q], WoB, b_out,
                                                             out + (size_t)t * BDIM * DDIM);
    }
}
